// GraphAttentionLayer_2327872274804
// MI455X (gfx1250) — compile-verified, matrix operand order checked by reading, not run on hardware
//
#include <hip/hip_runtime.h>
#include <hip/hip_bf16.h>
#include <math.h>

#define B_    16
#define N_    2048
#define FIN   256
#define FOUT  256
#define RELD  10
#define NREL  8
#define ALPHA 0.2f
#define LNEPS 1e-5f
#define NEGINF -9.0e15f

typedef __attribute__((ext_vector_type(16))) __bf16 v16bf;
typedef __attribute__((ext_vector_type(8)))  float  v8f;

union BFrag {
    v16bf v;
    unsigned short u[16];
    unsigned int   w[8];
};

// Hardware f32 -> bf16 (lowers to v_cvt_pk_bf16_f32 when pairs fuse)
__device__ inline unsigned short f2bf(float f) {
    return __builtin_bit_cast(unsigned short, (__bf16)f);
}
__device__ inline unsigned int f2bf2(float lo, float hi) {
    return (unsigned int)f2bf(lo) | ((unsigned int)f2bf(hi) << 16);
}

// k index inside a 16x32 bf16 A/B fragment for vgpr v, sub-element s (ISA 7.12.2):
// k = 16*(v/4) + 8*(lane/16) + 2*(v%4) + s
__device__ inline int frag_k(int v, int half) {
    return ((v >> 2) << 4) + (half << 3) + ((v & 3) << 1);
}

// ---------------------------------------------------------------------------
// Kernel 0: pack W (256x256 f32, row=k, col=n) into WT16[n][k] bf16 so GEMM-1
// B-fragment K-pairs are contiguous 32-bit loads.
// ---------------------------------------------------------------------------
__global__ __launch_bounds__(256) void pack_w_kernel(
    const float* __restrict__ W, unsigned short* __restrict__ WT16)
{
    const int n = blockIdx.x;              // 0..255
    const int k = threadIdx.x;             // 0..255
    WT16[(n << 8) | k] = f2bf(W[(k << 8) | n]);
}

// ---------------------------------------------------------------------------
// Kernel 1: Wh = h @ W  (bf16 WMMA, f32 accum). Writes Wh (f32, row major,
// for the a1/a2 dots) and WhT16 (bf16, [b][f][node], B operand of GEMM-2).
// Block: 256 thr = 8 waves (2 M-tiles x 4 N-tiles) -> 32 rows x 256 cols.
// ---------------------------------------------------------------------------
__global__ __launch_bounds__(256) void wh_gemm_kernel(
    const float* __restrict__ h, const unsigned short* __restrict__ WT16,
    float* __restrict__ Wh, unsigned short* __restrict__ WhT16)
{
    const int t    = threadIdx.x;
    const int lane = t & 31;
    const int wid  = t >> 5;
    const int wrow = wid >> 2;            // 0..1
    const int wcol = wid & 3;             // 0..3
    const int half = lane >> 4;
    const int l16  = lane & 15;

    const int rowBase = blockIdx.x * 32 + wrow * 16;   // flat row = b*N + i
    const int colBase = wcol * 64;

    v8f acc[4] = {};
    for (int k0 = 0; k0 < FIN; k0 += 32) {
        BFrag a;
        const float* hrow = h + (size_t)(rowBase + l16) * FIN;
        #pragma unroll
        for (int v = 0; v < 8; ++v) {
            const int k = k0 + frag_k(v, half);
            a.w[v] = f2bf2(hrow[k], hrow[k + 1]);      // v_cvt_pk_bf16_f32
        }
        #pragma unroll
        for (int c = 0; c < 4; ++c) {
            BFrag bb;
            const int n = colBase + c * 16 + l16;
            const unsigned short* wc = WT16 + ((size_t)n << 8);
            #pragma unroll
            for (int v = 0; v < 8; ++v) {
                const int k = k0 + frag_k(v, half);
                bb.w[v] = *(const unsigned int*)(wc + k);   // bf16 pair
            }
            acc[c] = __builtin_amdgcn_wmma_f32_16x16x32_bf16(
                false, a.v, false, bb.v, (short)0, acc[c], false, false);
        }
    }

    const int b = rowBase / N_;
    #pragma unroll
    for (int c = 0; c < 4; ++c) {
        #pragma unroll
        for (int r = 0; r < 8; ++r) {
            const int row = rowBase + r + (half << 3);
            const int col = colBase + c * 16 + l16;
            const float val = acc[c][r];
            Wh[(size_t)row * FOUT + col] = val;
            WhT16[((size_t)b * FOUT + col) * N_ + (row - b * N_)] = f2bf(val);
        }
    }
}

// ---------------------------------------------------------------------------
// Kernel 2: s_src / s_dst — one wave per node, both dots in one Wh pass.
// ---------------------------------------------------------------------------
__global__ __launch_bounds__(256) void score_vec_kernel(
    const float* __restrict__ Wh, const float* __restrict__ a,
    float* __restrict__ s_src, float* __restrict__ s_dst)
{
    const int node = blockIdx.x * 8 + (threadIdx.x >> 5);
    const int lane = threadIdx.x & 31;
    const float* row = Wh + (size_t)node * FOUT;
    float d1 = 0.f, d2 = 0.f;
    for (int f = lane; f < FOUT; f += 32) {
        const float w = row[f];
        d1 += w * a[f];
        d2 += w * a[FOUT + f];
    }
    #pragma unroll
    for (int off = 16; off > 0; off >>= 1) {
        d1 += __shfl_xor(d1, off, 32);
        d2 += __shfl_xor(d2, off, 32);
    }
    if (lane == 0) { s_src[node] = d1; s_dst[node] = d2; }
}

// rel_score[r] = rel_emb[r] . a3   (8-entry LUT)
__global__ void rel_score_kernel(const float* __restrict__ rel_emb,
                                 const float* __restrict__ a,
                                 float* __restrict__ rel_score)
{
    const int r = threadIdx.x;
    if (r < NREL) {
        float s = 0.f;
        for (int d = 0; d < RELD; ++d) s += rel_emb[r * RELD + d] * a[2 * FOUT + d];
        rel_score[r] = s;
    }
}

// ---------------------------------------------------------------------------
// Kernel 3: one block per (b,i) row. Whole 2048-wide row lives in registers
// (8 per thread). Single adj read; both softmaxes; writes attention_score
// (f32, model output) and masked attention (bf16, GEMM-2 A operand).
// Reductions: wave32 shfl_xor (5 steps) + 8-entry LDS combine.
// ---------------------------------------------------------------------------
__global__ __launch_bounds__(256) void attn_row_kernel(
    const int* __restrict__ adj, const float* __restrict__ s_src,
    const float* __restrict__ s_dst, const float* __restrict__ rel_score,
    float* __restrict__ score_out, unsigned short* __restrict__ attn16)
{
    __shared__ float sA[8];
    __shared__ float sB[8];
    __shared__ float rels[NREL];
    const int t    = threadIdx.x;
    const int lane = t & 31;
    const int wid  = t >> 5;
    const int row  = blockIdx.x;           // flat b*N + i
    const int b    = row >> 11;            // / 2048
    if (t < NREL) rels[t] = rel_score[t];
    __syncthreads();

    const float si = s_src[row];
    const int*   adjrow = adj   + (size_t)row * N_;
    const float* sd     = s_dst + (size_t)b * N_;

    float e[8], em[8];
    float maxu = -INFINITY, maxm = -INFINITY;
    #pragma unroll
    for (int k = 0; k < 8; ++k) {
        const int j  = t + (k << 8);
        const int av = adjrow[j];
        float x = si + sd[j] + rels[av];
        x = (x > 0.f) ? x : ALPHA * x;                 // leaky_relu
        e[k]  = x;
        em[k] = (av > 0) ? x : NEGINF;
        maxu = fmaxf(maxu, x);
        maxm = fmaxf(maxm, em[k]);
    }
    // block max: wave shuffle reduce, then 8-wave LDS combine
    #pragma unroll
    for (int off = 16; off > 0; off >>= 1) {
        maxu = fmaxf(maxu, __shfl_xor(maxu, off, 32));
        maxm = fmaxf(maxm, __shfl_xor(maxm, off, 32));
    }
    if (lane == 0) { sA[wid] = maxu; sB[wid] = maxm; }
    __syncthreads();
    maxu = sA[0]; maxm = sB[0];
    #pragma unroll
    for (int w = 1; w < 8; ++w) {
        maxu = fmaxf(maxu, sA[w]);
        maxm = fmaxf(maxm, sB[w]);
    }
    __syncthreads();

    float su = 0.f, sm = 0.f;
    float pu[8], pm[8];
    #pragma unroll
    for (int k = 0; k < 8; ++k) {
        pu[k] = expf(e[k]  - maxu);
        pm[k] = expf(em[k] - maxm);
        su += pu[k]; sm += pm[k];
    }
    #pragma unroll
    for (int off = 16; off > 0; off >>= 1) {
        su += __shfl_xor(su, off, 32);
        sm += __shfl_xor(sm, off, 32);
    }
    if (lane == 0) { sA[wid] = su; sB[wid] = sm; }
    __syncthreads();
    su = 0.f; sm = 0.f;
    #pragma unroll
    for (int w = 0; w < 8; ++w) { su += sA[w]; sm += sB[w]; }

    const float inv_u = 1.f / su;
    const float inv_m = 1.f / sm;

    float*          srow = score_out + (size_t)row * N_;
    unsigned short* arow = attn16    + (size_t)row * N_;
    #pragma unroll
    for (int k = 0; k < 8; ++k) {
        const int j = t + (k << 8);
        srow[j] = pu[k] * inv_u;
        arow[j] = f2bf(pm[k] * inv_m);
    }
}

// ---------------------------------------------------------------------------
// Kernel 4: h_prime = attention @ Wh (bf16 WMMA, K=2048), fused LN + GELU.
// Block = 8 waves (2x4) -> 32 rows x 256 cols, 32 KB LDS staging tile.
// ---------------------------------------------------------------------------
__global__ __launch_bounds__(256) void hprime_kernel(
    const unsigned short* __restrict__ attn16,
    const unsigned short* __restrict__ WhT16,
    const float* __restrict__ gamma, const float* __restrict__ beta,
    float* __restrict__ out)
{
    __shared__ float tile[32][FOUT];       // 32 KB
    __shared__ float mu_s[32], rs_s[32];
    const int t    = threadIdx.x;
    const int lane = t & 31;
    const int wid  = t >> 5;
    const int wrow = wid >> 2, wcol = wid & 3;
    const int half = lane >> 4, l16 = lane & 15;

    const int blocksPerB = N_ / 32;        // 64
    const int b     = blockIdx.x / blocksPerB;
    const int ibase = (blockIdx.x % blocksPerB) * 32;
    const int rowBase = ibase + wrow * 16;
    const int colBase = wcol * 64;

    const unsigned short* arow    = attn16 + ((size_t)b * N_ + rowBase + l16) * N_;
    const unsigned short* wt_base = WhT16  + (size_t)b * FOUT * N_;

    v8f acc[4] = {};
    for (int k0 = 0; k0 < N_; k0 += 32) {
        if (k0 + 32 < N_)
            __builtin_prefetch((const void*)(arow + k0 + 32), 0, 1);  // global_prefetch
        BFrag a;
        #pragma unroll
        for (int v = 0; v < 8; ++v) {
            const int k = k0 + frag_k(v, half);
            a.w[v] = *(const unsigned int*)(arow + k);      // bf16 pair
        }
        #pragma unroll
        for (int c = 0; c < 4; ++c) {
            BFrag bb;
            const int n = colBase + c * 16 + l16;
            const unsigned short* wc = wt_base + (size_t)n * N_;
            #pragma unroll
            for (int v = 0; v < 8; ++v) {
                const int k = k0 + frag_k(v, half);
                bb.w[v] = *(const unsigned int*)(wc + k);   // bf16 pair
            }
            acc[c] = __builtin_amdgcn_wmma_f32_16x16x32_bf16(
                false, a.v, false, bb.v, (short)0, acc[c], false, false);
        }
    }

    #pragma unroll
    for (int c = 0; c < 4; ++c)
        #pragma unroll
        for (int r = 0; r < 8; ++r)
            tile[wrow * 16 + r + (half << 3)][colBase + c * 16 + l16] = acc[c][r];
    __syncthreads();

    if (t < 32) {
        float s = 0.f, s2 = 0.f;
        for (int f = 0; f < FOUT; ++f) { const float x = tile[t][f]; s += x; s2 += x * x; }
        const float mu  = s / FOUT;
        const float var = s2 / FOUT - mu * mu;
        mu_s[t] = mu;
        rs_s[t] = rsqrtf(var + LNEPS);
    }
    __syncthreads();

    for (int idx = t; idx < 32 * FOUT; idx += 256) {
        const int r = idx >> 8, f = idx & 255;
        const float x = (tile[r][f] - mu_s[r]) * rs_s[r] * gamma[f] + beta[f];
        const float g = 0.5f * x * (1.f + erff(x * 0.70710678118654752f));  // exact GELU
        out[(((size_t)b * N_) + ibase + r) * FOUT + f] = g;
    }
}

// ---------------------------------------------------------------------------
extern "C" void kernel_launch(void* const* d_in, const int* in_sizes, int n_in,
                              void* d_out, int out_size, void* d_ws, size_t ws_size,
                              hipStream_t stream)
{
    (void)in_sizes; (void)n_in; (void)out_size; (void)ws_size;
    const float* h     = (const float*)d_in[0];
    const int*   adj   = (const int*)  d_in[1];
    const float* W     = (const float*)d_in[2];
    const float* a     = (const float*)d_in[3];
    const float* rel   = (const float*)d_in[4];
    const float* gamma = (const float*)d_in[5];
    const float* beta  = (const float*)d_in[6];

    float* out_f     = (float*)d_out;                       // (B,N,FOUT)
    float* score_out = out_f + (size_t)B_ * N_ * FOUT;      // (B,N,N)

    char* ws = (char*)d_ws;
    size_t off = 0;
    float*          Wh     = (float*)(ws + off);          off += (size_t)B_ * N_ * FOUT * 4;
    unsigned short* WhT16  = (unsigned short*)(ws + off); off += (size_t)B_ * N_ * FOUT * 2;
    unsigned short* attn16 = (unsigned short*)(ws + off); off += (size_t)B_ * N_ * N_ * 2;
    float*          s_src  = (float*)(ws + off);          off += (size_t)B_ * N_ * 4;
    float*          s_dst  = (float*)(ws + off);          off += (size_t)B_ * N_ * 4;
    float*          rels   = (float*)(ws + off);          off += 64;
    unsigned short* WT16   = (unsigned short*)(ws + off); off += (size_t)FIN * FOUT * 2;

    pack_w_kernel   <<<FOUT, FIN, 0, stream>>>(W, WT16);
    wh_gemm_kernel  <<<(B_ * N_) / 32, 256, 0, stream>>>(h, WT16, Wh, WhT16);
    score_vec_kernel<<<(B_ * N_) / 8, 256, 0, stream>>>(Wh, a, s_src, s_dst);
    rel_score_kernel<<<1, 32, 0, stream>>>(rel, a, rels);
    attn_row_kernel <<<B_ * N_, 256, 0, stream>>>(adj, s_src, s_dst, rels, score_out, attn16);
    hprime_kernel   <<<(B_ * N_) / 32, 256, 0, stream>>>(attn16, WhT16, gamma, beta, out_f);
}
